// AttLayer_12146167513544
// MI455X (gfx1250) — compile-verified, matrix operand order checked by reading, not run on hardware
//
#include <hip/hip_runtime.h>
#include <hip/hip_bf16.h>
#include <math.h>

typedef __attribute__((ext_vector_type(16))) __bf16 v16bf;
typedef __attribute__((ext_vector_type(8)))  __bf16 v8bf;
typedef __attribute__((ext_vector_type(8)))  float  v8f;

#define B_DIM   256
#define S_DIM   512
#define D_DIM   256
#define ATT_DIM 128

// ---------------------------------------------------------------------------
// Kernel 1: e[b*S+s] = exp( sum_n u[n] * tanh( (x[b,s,:] @ W)[n] + bias[n] ) )
// GEMM done with v_wmma_f32_16x16x32_bf16, M = B*S flattened rows.
// Each wave: 16 rows x 128 cols. 256 threads = 8 waves = 128 rows per block.
// ---------------------------------------------------------------------------
__launch_bounds__(256, 1)
__global__ void att_scores_wmma(const float* __restrict__ x,
                                const float* __restrict__ W,
                                const float* __restrict__ bias,
                                const float* __restrict__ u,
                                float* __restrict__ escore)
{
    // W transposed into LDS as bf16: Wt[n][k], n in [0,128), k in [0,256). 64 KB.
    __shared__ __bf16 Wt[ATT_DIM * D_DIM];

    const int tid = threadIdx.x;
    for (int i = tid; i < ATT_DIM * D_DIM; i += 256) {
        const int k = i >> 7;              // row of W (D index)
        const int n = i & (ATT_DIM - 1);   // col of W (ATT index), coalesced in global
        Wt[n * D_DIM + k] = (__bf16)W[i];
    }
    __syncthreads();

    const int wave = tid >> 5;
    const int lane = tid & 31;
    const int l16  = lane & 15;
    const int hi   = lane >> 4;            // 0: lanes 0-15, 1: lanes 16-31
    const int kofs = hi ? 8 : 0;           // per-lane K base offset (16-bit A/B layout)

    const long mbase = ((long)blockIdx.x * 8 + wave) * 16;  // first row of this wave's tile

    // Per-lane bias/u values: lane l16 covers column n = 16*nt + l16 for each n-tile nt.
    float bv[8], uv[8];
#pragma unroll
    for (int t = 0; t < 8; ++t) {
        bv[t] = bias[l16 + 16 * t];
        uv[t] = u[l16 + 16 * t];
    }

    // Load all 8 A-fragments (full K=256) for this wave's 16 rows; x read once.
    // 16-bit A 16x32 layout: lane holds K = kofs..kofs+7 then kofs+16..kofs+23.
    const float* xrow = x + (mbase + l16) * (long)D_DIM;
    v16bf A[8];
#pragma unroll
    for (int ks = 0; ks < 8; ++ks) {
        const int kb = ks * 32 + kofs;
        union { v16bf v; __bf16 w[16]; } af;
#pragma unroll
        for (int j = 0; j < 8; ++j) af.w[j]     = (__bf16)xrow[kb + j];
#pragma unroll
        for (int j = 0; j < 8; ++j) af.w[8 + j] = (__bf16)xrow[kb + 16 + j];
        A[ks] = af.v;
    }

    // Per-lane partial score: C-layout => acc[r] on this lane is row (r + 8*hi),
    // column n = 16*nt + l16. Accumulate tanh(.)*u over all n-tiles.
    float part[8];
#pragma unroll
    for (int r = 0; r < 8; ++r) part[r] = 0.0f;

#pragma unroll
    for (int nt = 0; nt < 8; ++nt) {
        const __bf16* wcol = &Wt[(nt * 16 + l16) * D_DIM];
        v8f acc = {};
#pragma unroll
        for (int ks = 0; ks < 8; ++ks) {
            const int kb = ks * 32 + kofs;
            union { v16bf v; v8bf h[2]; } bf;
            bf.h[0] = *(const v8bf*)(wcol + kb);        // K = kb .. kb+7      (16B ds load)
            bf.h[1] = *(const v8bf*)(wcol + kb + 16);   // K = kb+16 .. kb+23  (16B ds load)
            acc = __builtin_amdgcn_wmma_f32_16x16x32_bf16(
                      false, A[ks], false, bf.v, (short)0, acc, false, false);
        }
#pragma unroll
        for (int r = 0; r < 8; ++r)
            part[r] += tanhf(acc[r] + bv[nt]) * uv[nt];
    }

    // Reduce over the 16 lanes of each half-wave (columns) -> per-row score.
#pragma unroll
    for (int r = 0; r < 8; ++r) {
        float v = part[r];
        v += __shfl_xor(v, 1, 32);
        v += __shfl_xor(v, 2, 32);
        v += __shfl_xor(v, 4, 32);
        v += __shfl_xor(v, 8, 32);
        part[r] = v;
    }
    if (l16 == 0) {
        const long base = mbase + (hi ? 8 : 0);
#pragma unroll
        for (int r = 0; r < 8; ++r)
            escore[base + r] = __expf(part[r]);
    }
}

// ---------------------------------------------------------------------------
// Kernel 2: out[b,d] = (1/(sum_s e[b,s] + 1e-8)) * sum_s x[b,s,d] * e[b,s]
// One block per batch, thread t owns output column d = t. Coalesced streaming.
// ---------------------------------------------------------------------------
__launch_bounds__(256, 1)
__global__ void att_weighted_sum(const float* __restrict__ x,
                                 const float* __restrict__ escore,
                                 float* __restrict__ out)
{
    __shared__ float wsh[S_DIM];
    __shared__ float red[8];

    const int b   = blockIdx.x;
    const int tid = threadIdx.x;

    const float s0 = escore[(long)b * S_DIM + tid];
    const float s1 = escore[(long)b * S_DIM + 256 + tid];
    wsh[tid]       = s0;
    wsh[tid + 256] = s1;

    float sum = s0 + s1;
#pragma unroll
    for (int m = 1; m < 32; m <<= 1) sum += __shfl_xor(sum, m, 32);
    if ((tid & 31) == 0) red[tid >> 5] = sum;
    __syncthreads();

    float total = 0.0f;
#pragma unroll
    for (int i = 0; i < 8; ++i) total += red[i];
    const float inv = 1.0f / (total + 1e-8f);

    const float* xb = x + (long)b * S_DIM * D_DIM;
    float acc = 0.0f;
#pragma unroll 8
    for (int s = 0; s < S_DIM; ++s)
        acc = fmaf(xb[(long)s * D_DIM + tid], wsh[s], acc);

    out[(long)b * D_DIM + tid] = acc * inv;
}

// ---------------------------------------------------------------------------
extern "C" void kernel_launch(void* const* d_in, const int* in_sizes, int n_in,
                              void* d_out, int out_size, void* d_ws, size_t ws_size,
                              hipStream_t stream)
{
    const float* x  = (const float*)d_in[0];   // [B, S, D]
    const float* W  = (const float*)d_in[1];   // [D, ATT]
    const float* bb = (const float*)d_in[2];   // [ATT]
    const float* u  = (const float*)d_in[3];   // [ATT, 1]
    float* out = (float*)d_out;                // [B, D]
    float* esc = (float*)d_ws;                 // [B*S] scratch scores (512 KB)

    // 131072 rows total / (8 waves * 16 rows) = 1024 blocks
    const int nblk = (B_DIM * S_DIM) / (8 * 16);
    att_scores_wmma<<<nblk, 256, 0, stream>>>(x, W, bb, u, esc);
    att_weighted_sum<<<B_DIM, 256, 0, stream>>>(x, esc, out);
}
